// MolecularVAE_12051678232910
// MI455X (gfx1250) — matrix operand order repaired and checked by reading, compiled, not run on hardware
//
#include <hip/hip_runtime.h>
#include <cstdint>

#define DEV __device__ __forceinline__

typedef __attribute__((ext_vector_type(16))) __bf16 v16bf;
typedef __attribute__((ext_vector_type(8)))  __bf16 v8bf;
typedef __attribute__((ext_vector_type(8)))  float  v8f;
typedef __attribute__((ext_vector_type(4)))  unsigned int u32x4;
typedef __attribute__((ext_vector_type(8)))  int          i32x8;
typedef __attribute__((ext_vector_type(4)))  int          i32x4;

// ---------------- dimensions ----------------
constexpr int Bsz = 1024, T = 120, V = 53, Z = 292, H = 501;
constexpr int Hp = 512;          // padded hidden
constexpr int Zp = 320;          // padded z dim
constexpr int NG = 3 * Hp;       // 1536 padded gate columns
constexpr int Vp = 64;           // padded vocab
constexpr int KT_Z = Zp / 32;    // 10 k-tiles
constexpr int KT_H = Hp / 32;    // 16 k-tiles
constexpr int LDS_STRIDE = 520;  // Hp + 4 dwords TDM padding -> conflict-free ds reads

// ---------------- small helpers ----------------
DEV v8f vzero() {
  v8f z;
#pragma unroll
  for (int i = 0; i < 8; ++i) z[i] = 0.f;
  return z;
}

DEV float selu(float x) {
  const float scale = 1.0507009873554805f, alpha = 1.6732632423543772f;
  return x > 0.f ? scale * x : scale * alpha * (__expf(x) - 1.f);
}
DEV float sigmoidf(float x) { return 1.f / (1.f + __expf(-x)); }

DEV v8f wmma_bf16(v16bf a, v16bf b, v8f c) {
  return __builtin_amdgcn_wmma_f32_16x16x32_bf16(false, a, false, b, (short)0, c,
                                                 false, false);
}

// A fragment from row-major global memory
DEV v16bf load_a_frag(const __bf16* __restrict__ A, int lda, int mbase, int kb, int lane) {
  int row  = mbase + (lane & 15);
  int koff = (lane >> 4) * 8;
  const __bf16* p = A + (size_t)row * lda + kb + koff;
  union { v16bf v; v8bf h[2]; } u;
  u.h[0] = *(const v8bf*)p;
  u.h[1] = *(const v8bf*)(p + 16);
  return u.v;
}

// A fragment from LDS staging buffer (row stride LDS_STRIDE bf16)
DEV v16bf load_a_frag_lds(const __bf16* S, int mrel, int kb, int lane) {
  int row  = mrel + (lane & 15);
  int koff = (lane >> 4) * 8;
  const __bf16* p = S + row * LDS_STRIDE + kb + koff;
  union { v16bf v; v8bf h[2]; } u;
  u.h[0] = *(const v8bf*)p;
  u.h[1] = *(const v8bf*)(p + 16);
  return u.v;
}

// B fragment from packed layout Bp[ktile][Np][32]
DEV v16bf load_b_frag(const __bf16* __restrict__ Bp, int ktile, int Np, int col, int lane) {
  int c   = col + (lane & 15);
  int kof = (lane >> 4) * 8;
  const __bf16* p = Bp + ((size_t)ktile * Np + c) * 32 + kof;
  union { v16bf v; v8bf h[2]; } u;
  u.h[0] = *(const v8bf*)p;
  u.h[1] = *(const v8bf*)(p + 16);
  return u.v;
}

// Issue a TDM 2-D tile load: [rows x Hp] bf16 tile from row-major tensor (stride lda
// elements) into LDS at lds_off, with 16B LDS padding per 1024B row (-> LDS_STRIDE).
// Must be executed by a single wave; caller syncs with s_wait_tensorcnt + barrier.
DEV void tdm_load_tile(const __bf16* gsrc, int lda, unsigned lds_off, int rows) {
  unsigned long long ga = (unsigned long long)(uintptr_t)gsrc;
  u32x4 g0;
  g0[0] = 1u;                                   // count=1, user descriptor
  g0[1] = lds_off;                              // lds_addr (bytes)
  g0[2] = (unsigned)(ga & 0xFFFFFFFFu);         // global_addr[31:0]
  g0[3] = (unsigned)((ga >> 32) & 0x01FFFFFFu)  // global_addr[56:32]
          | 0x80000000u;                        // type=2 ("image")
  i32x8 g1;
  g1[0] = (int)((1u << 16)                      // data_size = 2 bytes
              | (1u << 20)                      // pad_enable
              | (7u << 22)                      // pad_interval: 256 dwords (=1024B row)
              | (3u << 25));                    // pad_amount: 4 dwords (=16B)
  unsigned td0 = 1u << 30, td1 = 1u << 20;      // large tensor dims (no OOB clipping)
  g1[1] = (int)((td0 & 0xFFFFu) << 16);                       // tensor_dim0 lo
  g1[2] = (int)(((td0 >> 16) & 0xFFFFu) | ((td1 & 0xFFFFu) << 16));
  g1[3] = (int)(((td1 >> 16) & 0xFFFFu) | ((unsigned)Hp << 16)); // tile_dim0 = Hp
  g1[4] = rows & 0xFFFF;                        // tile_dim1 = rows, tile_dim2 = 0
  g1[5] = lda;                                  // tensor_dim0_stride[31:0] (elements)
  g1[6] = 0;                                    // stride hi, dim1_stride lo
  g1[7] = 0;
  i32x4 gz4;
  gz4[0] = 0; gz4[1] = 0; gz4[2] = 0; gz4[3] = 0;   // 2-D: groups 2/3 unused
  i32x8 gz8;
#pragma unroll
  for (int i = 0; i < 8; ++i) gz8[i] = 0;
  __builtin_amdgcn_tensor_load_to_lds(g0, g1, gz4, gz4, gz8, 0);
}

// ---------------- encoder: conv stack + linears + reparam ----------------
__global__ void encoder_kernel(
    const float* __restrict__ x, const float* __restrict__ eps,
    const float* __restrict__ c1w, const float* __restrict__ c1b,
    const float* __restrict__ c2w, const float* __restrict__ c2b,
    const float* __restrict__ c3w, const float* __restrict__ c3b,
    const float* __restrict__ l0w, const float* __restrict__ l0b,
    const float* __restrict__ l1w, const float* __restrict__ l1b,
    const float* __restrict__ l2w, const float* __restrict__ l2b,
    const float* __restrict__ l3w, const float* __restrict__ l3b,
    float* __restrict__ zmean_out, float* __restrict__ zlogv_out,
    __bf16* __restrict__ zd_out) {
  __shared__ float sx[120 * 53];
  __shared__ float s1[9 * 45];
  __shared__ float s2[9 * 37];
  __shared__ float s3[270];
  __shared__ float s4[435];
  __shared__ float sz[292];
  const int b = blockIdx.x, tid = threadIdx.x;

  for (int i = tid; i < 120 * 53; i += 128) sx[i] = x[(size_t)b * (120 * 53) + i];
  __syncthreads();
  for (int j = tid; j < 9 * 45; j += 128) {
    int o = j / 45, l = j % 45;
    float a = c1b[o];
    for (int i = 0; i < 120; ++i) {
      const float* wp = &c1w[(o * 120 + i) * 9];
      const float* xp = &sx[i * 53 + l];
#pragma unroll
      for (int k = 0; k < 9; ++k) a += xp[k] * wp[k];
    }
    s1[j] = selu(a);
  }
  __syncthreads();
  for (int j = tid; j < 9 * 37; j += 128) {
    int o = j / 37, l = j % 37;
    float a = c2b[o];
    for (int i = 0; i < 9; ++i) {
      const float* wp = &c2w[(o * 9 + i) * 9];
      const float* xp = &s1[i * 45 + l];
#pragma unroll
      for (int k = 0; k < 9; ++k) a += xp[k] * wp[k];
    }
    s2[j] = selu(a);
  }
  __syncthreads();
  for (int j = tid; j < 270; j += 128) {
    int o = j / 27, l = j % 27;
    float a = c3b[o];
    for (int i = 0; i < 9; ++i) {
      const float* wp = &c3w[(o * 9 + i) * 11];
      const float* xp = &s2[i * 37 + l];
#pragma unroll
      for (int k = 0; k < 11; ++k) a += xp[k] * wp[k];
    }
    s3[j] = selu(a);
  }
  __syncthreads();
  for (int j = tid; j < 435; j += 128) {
    float a = l0b[j];
    const float* wp = &l0w[(size_t)j * 270];
    for (int i = 0; i < 270; ++i) a += s3[i] * wp[i];
    s4[j] = selu(a);
  }
  __syncthreads();
  for (int j = tid; j < Z; j += 128) {
    float m = l1b[j], lv = l2b[j];
    const float* w1p = &l1w[(size_t)j * 435];
    const float* w2p = &l2w[(size_t)j * 435];
    for (int i = 0; i < 435; ++i) { m += s4[i] * w1p[i]; lv += s4[i] * w2p[i]; }
    zmean_out[(size_t)b * Z + j] = m;
    zlogv_out[(size_t)b * Z + j] = lv;
    sz[j] = __expf(0.5f * lv) * (0.01f * eps[(size_t)b * Z + j]) + m;
  }
  __syncthreads();
  for (int j = tid; j < Zp; j += 128) {
    float d = 0.f;
    if (j < Z) {
      float a = l3b[j];
      const float* wp = &l3w[(size_t)j * Z];
      for (int i = 0; i < Z; ++i) a += sz[i] * wp[i];
      d = selu(a);
    }
    zd_out[(size_t)b * Zp + j] = (__bf16)d;
  }
}

// ---------------- weight packing ----------------
__global__ void pack_gate_w(const float* __restrict__ W, __bf16* __restrict__ dst,
                            int Kreal, int ktiles, int kt_off) {
  int total = ktiles * NG * 32;
  for (int idx = blockIdx.x * blockDim.x + threadIdx.x; idx < total;
       idx += gridDim.x * blockDim.x) {
    int kk = idx & 31;
    int n  = (idx >> 5) % NG;
    int kt = idx / (32 * NG);
    int k  = kt * 32 + kk;
    int g = n >> 9, ni = n & (Hp - 1);
    float v = 0.f;
    if (ni < H && k < Kreal) v = W[(size_t)(g * H + ni) * Kreal + k];
    dst[((size_t)(kt_off + kt) * NG + n) * 32 + kk] = (__bf16)v;
  }
}

__global__ void pack_lin4(const float* __restrict__ W, __bf16* __restrict__ dst) {
  int total = KT_H * Vp * 32;
  for (int idx = blockIdx.x * blockDim.x + threadIdx.x; idx < total;
       idx += gridDim.x * blockDim.x) {
    int kk = idx & 31;
    int n  = (idx >> 5) % Vp;
    int kt = idx / (32 * Vp);
    int k  = kt * 32 + kk;
    float v = (n < V && k < H) ? W[(size_t)n * H + k] : 0.f;
    dst[((size_t)kt * Vp + n) * 32 + kk] = (__bf16)v;
  }
}

__global__ void prep_bias_layer(const float* __restrict__ bi, const float* __restrict__ bh,
                                float* __restrict__ dst, int layer0) {
  int c = blockIdx.x * blockDim.x + threadIdx.x;
  if (c >= Hp) return;
  float br = 0.f, bz = 0.f, bx = 0.f, bn = 0.f;
  if (c < H) {
    if (layer0) { br = bh[c]; bz = bh[H + c]; bx = 0.f; bn = bh[2 * H + c]; }
    else { br = bi[c] + bh[c]; bz = bi[H + c] + bh[H + c]; bx = bi[2 * H + c]; bn = bh[2 * H + c]; }
  }
  dst[c] = br; dst[Hp + c] = bz; dst[2 * Hp + c] = bx; dst[3 * Hp + c] = bn;
}

__global__ void prep_bias_xg0(const float* __restrict__ bi, float* __restrict__ dst) {
  int n = blockIdx.x * blockDim.x + threadIdx.x;
  if (n >= NG) return;
  int g = n >> 9, ni = n & (Hp - 1);
  dst[n] = (ni < H) ? bi[g * H + ni] : 0.f;
}

// ---------------- xg0 = zd @ wi0^T + bi0  (WMMA) ----------------
__global__ void wmma_gemm_xg0(const __bf16* __restrict__ A, const __bf16* __restrict__ Bp,
                              const float* __restrict__ bias, float* __restrict__ C) {
  const int lane = threadIdx.x & 31, wave = threadIdx.x >> 5;
  const int wm = wave >> 1, wn = wave & 1;
  const int Mb = blockIdx.x * 64 + wm * 32;
  const int Nb = blockIdx.y * 64 + wn * 32;
  v8f acc[2][2];
#pragma unroll
  for (int mi = 0; mi < 2; ++mi)
#pragma unroll
    for (int ni = 0; ni < 2; ++ni) acc[mi][ni] = vzero();

  for (int kt = 0; kt < KT_Z; ++kt) {
    v16bf a0 = load_a_frag(A, Zp, Mb, kt * 32, lane);
    v16bf a1 = load_a_frag(A, Zp, Mb + 16, kt * 32, lane);
    v16bf b0 = load_b_frag(Bp, kt, NG, Nb, lane);
    v16bf b1 = load_b_frag(Bp, kt, NG, Nb + 16, lane);
    acc[0][0] = wmma_bf16(a0, b0, acc[0][0]);
    acc[0][1] = wmma_bf16(a0, b1, acc[0][1]);
    acc[1][0] = wmma_bf16(a1, b0, acc[1][0]);
    acc[1][1] = wmma_bf16(a1, b1, acc[1][1]);
  }
  const int cl = lane & 15, rh = (lane >> 4) * 8;
#pragma unroll
  for (int mi = 0; mi < 2; ++mi)
#pragma unroll
    for (int ni = 0; ni < 2; ++ni) {
      int col = Nb + ni * 16 + cl;
      float bb = bias[col];
#pragma unroll
      for (int v = 0; v < 8; ++v)
        C[(size_t)(Mb + mi * 16 + rh + v) * NG + col] = acc[mi][ni][v] + bb;
    }
}

// ---------------- fused GRU step: TDM-staged A, WMMA GEMM, gate math -------------
__global__ void gru_step_kernel(
    const __bf16* __restrict__ xin, int ldx,           // input slice at time t (or null)
    const float* __restrict__ xg_pre,                  // precomputed xg0 (or null)
    const __bf16* __restrict__ hin,                    // [1024 x Hp]
    const __bf16* __restrict__ Bp, int ktiles_in, int ktiles_h,
    const float* __restrict__ bias,                    // [4*Hp]
    __bf16* __restrict__ hout,                         // [1024 x Hp]
    __bf16* __restrict__ out_t, int ldo) {             // out[:, t, :]
  // One staging tile, reused for the x-phase then the h-phase (TDM pads each
  // 1024B row by 16B -> LDS_STRIDE keeps ds reads bank-conflict free).
  __shared__ __align__(128) __bf16 sA[64 * LDS_STRIDE];
  const int lane = threadIdx.x & 31, wave = threadIdx.x >> 5;
  const int wm = wave >> 1, wn = wave & 1;
  const int Mb = blockIdx.x * 64;                // block row base
  const int Nb = blockIdx.y * 64 + wn * 32;      // column within a single gate [0,Hp)
  const int mrel0 = wm * 32;                     // wave's rows inside the staged tile
  const int cl = lane & 15, rh = (lane >> 4) * 8;
  const unsigned lds_off = (unsigned)(uintptr_t)(void*)&sA[0];

  // acc groups: 0=r, 1=z, 2=n_x, 3=n_h
  v8f acc[4][2][2];
#pragma unroll
  for (int g = 0; g < 4; ++g)
#pragma unroll
    for (int mi = 0; mi < 2; ++mi)
#pragma unroll
      for (int ni = 0; ni < 2; ++ni) acc[g][mi][ni] = vzero();

  if (xg_pre) {  // layer 0: broadcast input projection loaded straight into C layout
#pragma unroll
    for (int g = 0; g < 3; ++g)
#pragma unroll
      for (int mi = 0; mi < 2; ++mi)
#pragma unroll
        for (int ni = 0; ni < 2; ++ni) {
          int col = g * Hp + Nb + ni * 16 + cl;
#pragma unroll
          for (int v = 0; v < 8; ++v)
            acc[g][mi][ni][v] = xg_pre[(size_t)(Mb + mrel0 + mi * 16 + rh + v) * NG + col];
        }
  }

  // Phase 0: input projection (gates r, z, n_x)
  if (ktiles_in > 0) {
    if (wave == 0) {
      tdm_load_tile(xin + (size_t)Mb * ldx, ldx, lds_off, 64);
      __builtin_amdgcn_s_wait_tensorcnt(0);
    }
    __syncthreads();
    for (int kt = 0; kt < ktiles_in; ++kt) {
      v16bf a0 = load_a_frag_lds(sA, mrel0, kt * 32, lane);
      v16bf a1 = load_a_frag_lds(sA, mrel0 + 16, kt * 32, lane);
#pragma unroll
      for (int g = 0; g < 3; ++g) {
        v16bf b0 = load_b_frag(Bp, kt, NG, g * Hp + Nb, lane);
        v16bf b1 = load_b_frag(Bp, kt, NG, g * Hp + Nb + 16, lane);
        acc[g][0][0] = wmma_bf16(a0, b0, acc[g][0][0]);
        acc[g][0][1] = wmma_bf16(a0, b1, acc[g][0][1]);
        acc[g][1][0] = wmma_bf16(a1, b0, acc[g][1][0]);
        acc[g][1][1] = wmma_bf16(a1, b1, acc[g][1][1]);
      }
    }
    __syncthreads();  // phase-0 reads done before tile reuse
  }

  // Phase 1: hidden projection (gates r, z, n_h); Bp rows offset by ktiles_in
  if (wave == 0) {
    tdm_load_tile(hin + (size_t)Mb * Hp, Hp, lds_off, 64);
    __builtin_amdgcn_s_wait_tensorcnt(0);
  }
  __syncthreads();
  const int gmap[3] = {0, 1, 3};
  for (int kt = 0; kt < ktiles_h; ++kt) {
    v16bf a0 = load_a_frag_lds(sA, mrel0, kt * 32, lane);
    v16bf a1 = load_a_frag_lds(sA, mrel0 + 16, kt * 32, lane);
#pragma unroll
    for (int gg = 0; gg < 3; ++gg) {
      const int g = gmap[gg];
      v16bf b0 = load_b_frag(Bp, ktiles_in + kt, NG, gg * Hp + Nb, lane);
      v16bf b1 = load_b_frag(Bp, ktiles_in + kt, NG, gg * Hp + Nb + 16, lane);
      acc[g][0][0] = wmma_bf16(a0, b0, acc[g][0][0]);
      acc[g][0][1] = wmma_bf16(a0, b1, acc[g][0][1]);
      acc[g][1][0] = wmma_bf16(a1, b0, acc[g][1][0]);
      acc[g][1][1] = wmma_bf16(a1, b1, acc[g][1][1]);
    }
  }

  // Gate math; h_old re-read from the staged LDS tile (still holds h)
#pragma unroll
  for (int mi = 0; mi < 2; ++mi)
#pragma unroll
    for (int ni = 0; ni < 2; ++ni) {
      int col = Nb + ni * 16 + cl;
      bool valid = col < H;
      float br = bias[col], bz = bias[Hp + col];
      float bx = bias[2 * Hp + col], bn = bias[3 * Hp + col];
#pragma unroll
      for (int v = 0; v < 8; ++v) {
        int rrel = mrel0 + mi * 16 + rh + v;
        int row  = Mb + rrel;
        float r_ = sigmoidf(acc[0][mi][ni][v] + br);
        float z_ = sigmoidf(acc[1][mi][ni][v] + bz);
        float n_ = tanhf(acc[2][mi][ni][v] + bx + r_ * (acc[3][mi][ni][v] + bn));
        float hold = (float)sA[rrel * LDS_STRIDE + col];
        float hnew = valid ? ((1.f - z_) * n_ + z_ * hold) : 0.f;
        __bf16 hb = (__bf16)hnew;
        hout[(size_t)row * Hp + col] = hb;
        out_t[(size_t)row * ldo + col] = hb;
      }
    }
}

// ---------------- logits + softmax ----------------
__global__ void logits_softmax_kernel(const __bf16* __restrict__ A,
                                      const __bf16* __restrict__ L4p,
                                      const float* __restrict__ b4,
                                      float* __restrict__ y) {
  __shared__ float sl[64][65];
  const int lane = threadIdx.x & 31, wave = threadIdx.x >> 5;
  const int wm = wave >> 1, wn = wave & 1;
  const int Mb = blockIdx.x * 64 + wm * 32;
  const int Nb = wn * 32;
  v8f acc[2][2];
#pragma unroll
  for (int mi = 0; mi < 2; ++mi)
#pragma unroll
    for (int ni = 0; ni < 2; ++ni) acc[mi][ni] = vzero();

  for (int kt = 0; kt < KT_H; ++kt) {
    v16bf a0 = load_a_frag(A, Hp, Mb, kt * 32, lane);
    v16bf a1 = load_a_frag(A, Hp, Mb + 16, kt * 32, lane);
    v16bf b0 = load_b_frag(L4p, kt, Vp, Nb, lane);
    v16bf b1 = load_b_frag(L4p, kt, Vp, Nb + 16, lane);
    acc[0][0] = wmma_bf16(a0, b0, acc[0][0]);
    acc[0][1] = wmma_bf16(a0, b1, acc[0][1]);
    acc[1][0] = wmma_bf16(a1, b0, acc[1][0]);
    acc[1][1] = wmma_bf16(a1, b1, acc[1][1]);
  }
  const int cl = lane & 15, rh = (lane >> 4) * 8;
#pragma unroll
  for (int mi = 0; mi < 2; ++mi)
#pragma unroll
    for (int ni = 0; ni < 2; ++ni) {
      int col = Nb + ni * 16 + cl;
      int lr = wm * 32 + mi * 16 + rh;
#pragma unroll
      for (int v = 0; v < 8; ++v) sl[lr + v][col] = acc[mi][ni][v];
    }
  __syncthreads();
  int tid = threadIdx.x;
  if (tid < 64) {
    size_t grow = (size_t)blockIdx.x * 64 + tid;
    float m = -1e30f;
    for (int c = 0; c < V; ++c) { float x = sl[tid][c] + b4[c]; m = fmaxf(m, x); }
    float s = 0.f;
    for (int c = 0; c < V; ++c) s += __expf(sl[tid][c] + b4[c] - m);
    float inv = 1.f / s;
    for (int c = 0; c < V; ++c) y[grow * V + c] = __expf(sl[tid][c] + b4[c] - m) * inv;
  }
}

// ---------------- host launch ----------------
extern "C" void kernel_launch(void* const* d_in, const int* in_sizes, int n_in,
                              void* d_out, int out_size, void* d_ws, size_t ws_size,
                              hipStream_t stream) {
  (void)in_sizes; (void)n_in; (void)out_size; (void)ws_size;
  const float* x    = (const float*)d_in[0];
  const float* eps  = (const float*)d_in[2];
  const float* c1w  = (const float*)d_in[3];  const float* c1b = (const float*)d_in[4];
  const float* c2w  = (const float*)d_in[5];  const float* c2b = (const float*)d_in[6];
  const float* c3w  = (const float*)d_in[7];  const float* c3b = (const float*)d_in[8];
  const float* l0w  = (const float*)d_in[9];  const float* l0b = (const float*)d_in[10];
  const float* l1w  = (const float*)d_in[11]; const float* l1b = (const float*)d_in[12];
  const float* l2w  = (const float*)d_in[13]; const float* l2b = (const float*)d_in[14];
  const float* l3w  = (const float*)d_in[15]; const float* l3b = (const float*)d_in[16];
  const float* wi0  = (const float*)d_in[17]; const float* wh0 = (const float*)d_in[18];
  const float* bi0  = (const float*)d_in[19]; const float* bh0 = (const float*)d_in[20];
  const float* wi1  = (const float*)d_in[21]; const float* wh1 = (const float*)d_in[22];
  const float* bi1  = (const float*)d_in[23]; const float* bh1 = (const float*)d_in[24];
  const float* wi2  = (const float*)d_in[25]; const float* wh2 = (const float*)d_in[26];
  const float* bi2  = (const float*)d_in[27]; const float* bh2 = (const float*)d_in[28];
  const float* l4w  = (const float*)d_in[33]; const float* l4b = (const float*)d_in[34];
  // x_idx, grul_*, embed are dead code in the reference (DCE'd): skipped.

  float* out   = (float*)d_out;
  float* y     = out;                                   // [B,T,V]
  float* zmean = out + (size_t)Bsz * T * V;             // [B,Z]
  float* zlogv = zmean + (size_t)Bsz * Z;               // [B,Z]

  char* ws = (char*)d_ws;
  constexpr size_t OFF_ZD   = 0;
  constexpr size_t OFF_XG0  = OFF_ZD   + (size_t)Bsz * Zp * 2;
  constexpr size_t OFF_P0   = OFF_XG0  + (size_t)Bsz * NG * 4;
  constexpr size_t OFF_W0H  = OFF_P0   + (size_t)KT_Z * NG * 32 * 2;
  constexpr size_t OFF_W1   = OFF_W0H  + (size_t)KT_H * NG * 32 * 2;
  constexpr size_t OFF_W2   = OFF_W1   + (size_t)2 * KT_H * NG * 32 * 2;
  constexpr size_t OFF_L4   = OFF_W2   + (size_t)2 * KT_H * NG * 32 * 2;
  constexpr size_t OFF_B0   = OFF_L4   + (size_t)KT_H * Vp * 32 * 2;
  constexpr size_t OFF_BL0  = OFF_B0   + (size_t)NG * 4;
  constexpr size_t OFF_BL1  = OFF_BL0  + (size_t)4 * Hp * 4;
  constexpr size_t OFF_BL2  = OFF_BL1  + (size_t)4 * Hp * 4;
  constexpr size_t OFF_HA   = OFF_BL2  + (size_t)4 * Hp * 4;
  constexpr size_t OFF_HB   = OFF_HA   + (size_t)Bsz * Hp * 2;
  constexpr size_t OFF_OUTA = OFF_HB   + (size_t)Bsz * Hp * 2;
  constexpr size_t OFF_OUTB = OFF_OUTA + (size_t)Bsz * T * Hp * 2;

  __bf16* zd    = (__bf16*)(ws + OFF_ZD);
  float*  xg0   = (float*) (ws + OFF_XG0);
  __bf16* p0    = (__bf16*)(ws + OFF_P0);
  __bf16* w0h   = (__bf16*)(ws + OFF_W0H);
  __bf16* w1p   = (__bf16*)(ws + OFF_W1);
  __bf16* w2p   = (__bf16*)(ws + OFF_W2);
  __bf16* l4p   = (__bf16*)(ws + OFF_L4);
  float*  b0p   = (float*) (ws + OFF_B0);
  float*  bl0   = (float*) (ws + OFF_BL0);
  float*  bl1   = (float*) (ws + OFF_BL1);
  float*  bl2   = (float*) (ws + OFF_BL2);
  __bf16* hA    = (__bf16*)(ws + OFF_HA);
  __bf16* hB    = (__bf16*)(ws + OFF_HB);
  __bf16* outA  = (__bf16*)(ws + OFF_OUTA);
  __bf16* outB  = (__bf16*)(ws + OFF_OUTB);

  // 1) encoder
  hipLaunchKernelGGL(encoder_kernel, dim3(Bsz), dim3(128), 0, stream,
                     x, eps, c1w, c1b, c2w, c2b, c3w, c3b, l0w, l0b,
                     l1w, l1b, l2w, l2b, l3w, l3b, zmean, zlogv, zd);

  // 2) weight packing + biases
  hipLaunchKernelGGL(pack_gate_w, dim3(1024), dim3(256), 0, stream, wi0, p0,  Z, KT_Z, 0);
  hipLaunchKernelGGL(pack_gate_w, dim3(1024), dim3(256), 0, stream, wh0, w0h, H, KT_H, 0);
  hipLaunchKernelGGL(pack_gate_w, dim3(1024), dim3(256), 0, stream, wi1, w1p, H, KT_H, 0);
  hipLaunchKernelGGL(pack_gate_w, dim3(1024), dim3(256), 0, stream, wh1, w1p, H, KT_H, KT_H);
  hipLaunchKernelGGL(pack_gate_w, dim3(1024), dim3(256), 0, stream, wi2, w2p, H, KT_H, 0);
  hipLaunchKernelGGL(pack_gate_w, dim3(1024), dim3(256), 0, stream, wh2, w2p, H, KT_H, KT_H);
  hipLaunchKernelGGL(pack_lin4,   dim3(128),  dim3(256), 0, stream, l4w, l4p);
  hipLaunchKernelGGL(prep_bias_xg0,   dim3(6), dim3(256), 0, stream, bi0, b0p);
  hipLaunchKernelGGL(prep_bias_layer, dim3(2), dim3(256), 0, stream, bi0, bh0, bl0, 1);
  hipLaunchKernelGGL(prep_bias_layer, dim3(2), dim3(256), 0, stream, bi1, bh1, bl1, 0);
  hipLaunchKernelGGL(prep_bias_layer, dim3(2), dim3(256), 0, stream, bi2, bh2, bl2, 0);

  // 3) broadcast input projection for layer 0 (constant over time)
  hipLaunchKernelGGL(wmma_gemm_xg0, dim3(Bsz / 64, NG / 64), dim3(128), 0, stream,
                     zd, p0, b0p, xg0);

  const dim3 sgrid(Bsz / 64, Hp / 64), sblk(128);
  const int ldo = T * Hp;

  // 4) layer 0 scan: xg precomputed, hidden GEMM only
  (void)hipMemsetAsync(hA, 0, (size_t)Bsz * Hp * 2, stream);
  {
    __bf16 *hi = hA, *ho = hB;
    for (int t = 0; t < T; ++t) {
      hipLaunchKernelGGL(gru_step_kernel, sgrid, sblk, 0, stream,
                         (const __bf16*)nullptr, 0, xg0, hi, w0h, 0, KT_H, bl0,
                         ho, outA + (size_t)t * Hp, ldo);
      __bf16* tmp = hi; hi = ho; ho = tmp;
    }
  }
  // 5) layer 1 scan: fused input (outA slice) + hidden GEMM
  (void)hipMemsetAsync(hA, 0, (size_t)Bsz * Hp * 2, stream);
  {
    __bf16 *hi = hA, *ho = hB;
    for (int t = 0; t < T; ++t) {
      hipLaunchKernelGGL(gru_step_kernel, sgrid, sblk, 0, stream,
                         outA + (size_t)t * Hp, ldo, (const float*)nullptr, hi, w1p,
                         KT_H, KT_H, bl1, ho, outB + (size_t)t * Hp, ldo);
      __bf16* tmp = hi; hi = ho; ho = tmp;
    }
  }
  // 6) layer 2 scan: reads outB, writes outA (reuse)
  (void)hipMemsetAsync(hA, 0, (size_t)Bsz * Hp * 2, stream);
  {
    __bf16 *hi = hA, *ho = hB;
    for (int t = 0; t < T; ++t) {
      hipLaunchKernelGGL(gru_step_kernel, sgrid, sblk, 0, stream,
                         outB + (size_t)t * Hp, ldo, (const float*)nullptr, hi, w2p,
                         KT_H, KT_H, bl2, ho, outA + (size_t)t * Hp, ldo);
      __bf16* tmp = hi; hi = ho; ho = tmp;
    }
  }
  // 7) logits + softmax
  hipLaunchKernelGGL(logits_softmax_kernel, dim3((Bsz * T) / 64), dim3(128), 0, stream,
                     outA, l4p, l4b, y);
}
